// NetVLAD_15719580304428
// MI455X (gfx1250) — matrix operand order repaired and checked by reading, not compiled, not run on hardware
//
#include <hip/hip_runtime.h>

#define C_DIM 256
#define K_CL  64
#define P_TOT 1024
#define PT    64
#define NTILES (P_TOT / PT)
#define EPSF  1e-12f

typedef _Float16 v8h  __attribute__((ext_vector_type(8)));
typedef _Float16 v16h __attribute__((ext_vector_type(16)));
typedef float    v8f  __attribute__((ext_vector_type(8)));
typedef unsigned int u32x4 __attribute__((ext_vector_type(4)));
typedef int          i32x4 __attribute__((ext_vector_type(4)));
typedef int          i32x8 __attribute__((ext_vector_type(8)));

// Load a 16-half WMMA fragment as two b128 LDS loads: halves [0..7] at p,
// halves [8..15] at p+gap (gap=16 for both A- and B-fragments).
__device__ __forceinline__ v16h ld_frag(const _Float16* p, int gap) {
  v8h lo = *(const v8h*)(p);
  v8h hi = *(const v8h*)(p + gap);
  v16h r;
#pragma unroll
  for (int i = 0; i < 8; ++i) { r[i] = lo[i]; r[i + 8] = hi[i]; }
  return r;
}

__device__ __forceinline__ v8f wmma16(v16h a, v16h b, v8f c) {
  return __builtin_amdgcn_wmma_f32_16x16x32_f16(
      false, a, false, b, (short)0, c, false, false);
}

// TDM: 2D tile load (tile_dim1=256 rows of tile_dim0=64 fp32, row stride 1024)
// from global into LDS at lds_off; LDS receives [c][p_local], p contiguous.
// This toolchain exposes the 6-arg builtin: (g0, g1, g2, g3, g4, cpol).
__device__ __forceinline__ void tdm_load_tile(const float* gaddr, unsigned lds_off) {
  const unsigned long long ga = (unsigned long long)(uintptr_t)gaddr;
  u32x4 g0;
  g0[0] = 1u;                                   // count=1 (valid user descriptor)
  g0[1] = lds_off;                              // lds_addr (bytes)
  g0[2] = (unsigned)(ga & 0xFFFFFFFFu);         // global_addr[31:0]
  g0[3] = (unsigned)((ga >> 32) & 0x1FFFFFFu)   // global_addr[56:32]
        | (2u << 30);                           // type=2 ("image")
  i32x8 g1;
  g1[0] = 0x00020000;                           // data_size=2 (4 bytes/elem)
  g1[1] = (int)(1024u << 16);                   // tensor_dim0 = 1024 (bits 79:48)
  g1[2] = (int)(256u << 16);                    // tensor_dim1 = 256  (bits 111:80)
  g1[3] = (int)(64u << 16);                     // tile_dim0  = 64    (bits 127:112)
  g1[4] = 256;                                  // tile_dim1  = 256   (bits 143:128)
  g1[5] = 1024;                                 // tensor_dim0_stride (bits 207:160)
  g1[6] = 0;
  g1[7] = 0;
  i32x4 gz4 = {0, 0, 0, 0};
  i32x8 gz8 = {0, 0, 0, 0, 0, 0, 0, 0};
  __builtin_amdgcn_tensor_load_to_lds(g0, g1, gz4, gz4, gz8, 0);
}

// LDS byte offsets (total ~249 KB of the 320 KB WGP LDS)
#define OFF_W     0          // 32768  f16 [64][256]  conv weights
#define OFF_XRAW0 32768      // 65536  f32 [256][PT]  TDM staging buf 0
#define OFF_XRAW1 98304      // 65536  f32 [256][PT]  TDM staging buf 1
#define OFF_XP    163840     // 32768  f16 [PT][256]  normalized x, p-major
#define OFF_XC    196608     // 32768  f16 [256][PT]  normalized x, c-major
#define OFF_AGG   163840     // 65536  f32 [64][256]  (reuses XP+XC after tiles)
#define OFF_A     229376     // 8192   f16 [64][PT]   softmax assignments
#define OFF_LOGIT 237568     // 16384  f32 [PT][64]
#define OFF_NORM2 253952     // 256    f32 [64]
#define OFF_ASUM  254208     // 256    f32 [64]
#define OFF_B     254464     // 256    f32 [64]
#define OFF_SINV  254720     // 256    f32 [64]
#define OFF_G     254976     // 16
#define SMEM_SZ   254992

__global__ __launch_bounds__(256, 1) void netvlad_kernel(
    const float* __restrict__ xg, const float* __restrict__ wg,
    const float* __restrict__ bg, const float* __restrict__ cg,
    float* __restrict__ outg) {
  __shared__ __align__(16) unsigned char smem[SMEM_SZ];
  _Float16* wl   = (_Float16*)(smem + OFF_W);
  _Float16* xp   = (_Float16*)(smem + OFF_XP);
  _Float16* xc   = (_Float16*)(smem + OFF_XC);
  float*    aggl = (float*)   (smem + OFF_AGG);
  _Float16* al   = (_Float16*)(smem + OFF_A);
  float*    lg   = (float*)   (smem + OFF_LOGIT);
  float*    nrm2 = (float*)   (smem + OFF_NORM2);
  float*    asum = (float*)   (smem + OFF_ASUM);
  float*    bl   = (float*)   (smem + OFF_B);
  float*    sinv = (float*)   (smem + OFF_SINV);
  float*    gss  = (float*)   (smem + OFF_G);

  const int tid  = threadIdx.x;
  const int n    = blockIdx.x;
  const int wv   = tid >> 5;       // wave id 0..7 (wave32)
  const int lane = tid & 31;
  const int l16  = lane & 15;
  const int lh   = lane >> 4;      // half-wave select

  const float* xbase = xg + (size_t)n * (C_DIM * P_TOT);
  const unsigned xraw0 = (unsigned)(uintptr_t)(smem + OFF_XRAW0);
  const unsigned xraw1 = (unsigned)(uintptr_t)(smem + OFF_XRAW1);

  // Kick off TDM for tile 0 (wave 0 only; EXEC ignored by TDM, TENSORcnt is
  // per-wave) so the DMA overlaps the weight conversion below.
  if (wv == 0) tdm_load_tile(xbase, xraw0);

  // ---- one-time init: weights -> f16 LDS, bias, accumulator zeroing ----
  for (int i = tid; i < K_CL * C_DIM; i += 256) wl[i] = (_Float16)wg[i];
  if (tid < K_CL) { bl[tid] = bg[tid]; asum[tid] = 0.f; }
  if (tid == 0) gss[0] = 0.f;

  // wave work assignment
  const int pt  = wv >> 1;          // logits p-tile (0..3)
  const int kts = (wv & 1) * 2;     // logits k-tiles: kts, kts+1
  const int kt  = wv >> 1;          // agg k-tile (0..3)
  const int cb8 = (wv & 1) * 8;     // agg c-tiles cb8..cb8+7

  const v8f vzero = {0.f, 0.f, 0.f, 0.f, 0.f, 0.f, 0.f, 0.f};
  v8f acc[8];
#pragma unroll
  for (int j = 0; j < 8; ++j) acc[j] = vzero;

  const int pth = tid & 63;            // pixel within tile handled for norm
  const int cbt = (tid >> 6) << 6;     // channel block base (0/64/128/192)

  for (int t = 0; t < NTILES; ++t) {
    // Prefetch next tile's raw x via TDM into the other buffer, then wait for
    // the current tile (TENSORcnt is in-order: <=1 means older op complete).
    if (wv == 0) {
      if (t + 1 < NTILES) {
        tdm_load_tile(xbase + (t + 1) * PT, ((t + 1) & 1) ? xraw1 : xraw0);
        __builtin_amdgcn_s_wait_tensorcnt(1);
      } else {
        __builtin_amdgcn_s_wait_tensorcnt(0);
      }
    }
    if (tid < PT) nrm2[tid] = 0.f;
    __syncthreads();   // tile t staged in LDS + nrm2 cleared

    const float* xr = (const float*)(smem + ((t & 1) ? OFF_XRAW1 : OFF_XRAW0));

    // ---- pass 1: per-pixel channel sum-of-squares (from LDS staging) ----
    float s = 0.f;
#pragma unroll 8
    for (int c = 0; c < 64; ++c) {
      float v = xr[(cbt + c) * PT + pth];
      s += v * v;
    }
    atomicAdd(&nrm2[pth], s);
    __syncthreads();

    // ---- pass 2: normalize, write f16 in both layouts ----
    const float sc = 1.f / fmaxf(sqrtf(nrm2[pth]), EPSF);
#pragma unroll 8
    for (int c = 0; c < 64; ++c) {
      const int cc = cbt + c;
      _Float16 h = (_Float16)(xr[cc * PT + pth] * sc);
      xp[pth * C_DIM + cc] = h;
      xc[cc * PT + pth]    = h;
    }
    __syncthreads();

    // ---- logits GEMM: D[p,k] = x^T(p,c) * w^T(c,k), software-pipelined ----
    v8f d0 = vzero, d1 = vzero;
    v16h afA = ld_frag(xp + (pt * 16 + l16) * C_DIM + lh * 8, 16);
    v16h b0A = ld_frag(wl + (kts * 16 + l16) * C_DIM + lh * 8, 16);
    v16h b1A = ld_frag(wl + ((kts + 1) * 16 + l16) * C_DIM + lh * 8, 16);
#pragma unroll
    for (int cc = 0; cc < 8; ++cc) {
      v16h afN = afA, b0N = b0A, b1N = b1A;
      if (cc < 7) {   // prefetch next K-slice while current WMMAs execute
        afN = ld_frag(xp + (pt * 16 + l16) * C_DIM + (cc + 1) * 32 + lh * 8, 16);
        b0N = ld_frag(wl + (kts * 16 + l16) * C_DIM + (cc + 1) * 32 + lh * 8, 16);
        b1N = ld_frag(wl + ((kts + 1) * 16 + l16) * C_DIM + (cc + 1) * 32 + lh * 8, 16);
      }
      d0 = wmma16(afA, b0A, d0);
      d1 = wmma16(afA, b1A, d1);
      afA = afN; b0A = b0N; b1A = b1N;
    }
#pragma unroll
    for (int r = 0; r < 8; ++r) {   // D layout: M = r + 8*lh, N = l16
      const int prow = pt * 16 + r + lh * 8;
      lg[prow * K_CL + kts * 16 + l16]       = d0[r];
      lg[prow * K_CL + (kts + 1) * 16 + l16] = d1[r];
    }
    __syncthreads();

    // ---- fp32 softmax over K per pixel; a -> f16 [k][p] ----
    if (tid < PT) {
      float m = -3.4e38f;
      for (int k = 0; k < K_CL; ++k) m = fmaxf(m, lg[tid * K_CL + k] + bl[k]);
      float sum = 0.f;
      for (int k = 0; k < K_CL; ++k) {
        float e = __expf(lg[tid * K_CL + k] + bl[k] - m);
        lg[tid * K_CL + k] = e;
        sum += e;
      }
      const float inv = 1.f / sum;
      for (int k = 0; k < K_CL; ++k)
        al[k * PT + tid] = (_Float16)(lg[tid * K_CL + k] * inv);
    }
    __syncthreads();
    if (tid < K_CL) {  // asum[k] += sum_p a[k,p]
      float s2 = 0.f;
      for (int p = 0; p < PT; ++p) s2 += (float)al[tid * PT + p];
      asum[tid] += s2;
    }
    __syncthreads();

    // ---- agg GEMM: agg[k,c] += a(k,p)*x^T(p,c); batch loads, then WMMAs ----
#pragma unroll
    for (int pc = 0; pc < 2; ++pc) {
      v16h af = ld_frag(al + (kt * 16 + l16) * PT + pc * 32 + lh * 8, 16);
      v16h bf[8];
#pragma unroll
      for (int j = 0; j < 8; ++j)
        bf[j] = ld_frag(xc + ((cb8 + j) * 16 + l16) * PT + pc * 32 + lh * 8, 16);
#pragma unroll
      for (int j = 0; j < 8; ++j)
        acc[j] = wmma16(af, bf[j], acc[j]);
    }
    __syncthreads();
  }

  // ---- spill agg accumulators to LDS (reuses xp/xc region) ----
#pragma unroll
  for (int j = 0; j < 8; ++j)
#pragma unroll
    for (int r = 0; r < 8; ++r)
      aggl[(kt * 16 + r + lh * 8) * C_DIM + (cb8 + j) * 16 + l16] = acc[j][r];
  __syncthreads();

  // ---- intra-normalization scales + global norm accumulation ----
  if (tid < K_CL) {
    const float as = asum[tid];
    float ss = 0.f;
    for (int c = 0; c < C_DIM; ++c) {
      float v = aggl[tid * C_DIM + c] - as * cg[tid * C_DIM + c];
      ss += v * v;
    }
    const float m  = fmaxf(sqrtf(ss), EPSF);
    const float si = 1.f / m;
    sinv[tid] = si;
    atomicAdd(gss, ss * si * si);
  }
  __syncthreads();

  // ---- final write: vlad * intra_scale * global_scale (coalesced) ----
  const float gs = 1.f / fmaxf(sqrtf(gss[0]), EPSF);
  float* orow = outg + (size_t)n * (K_CL * C_DIM);
#pragma unroll 4
  for (int k = 0; k < K_CL; ++k) {
    const int e = k * C_DIM + tid;  // tid == channel c
    orow[e] = (aggl[e] - asum[k] * cg[e]) * sinv[k] * gs;
  }
}

extern "C" void kernel_launch(void* const* d_in, const int* in_sizes, int n_in,
                              void* d_out, int out_size, void* d_ws, size_t ws_size,
                              hipStream_t stream) {
  const float* x    = (const float*)d_in[0];
  const float* w    = (const float*)d_in[1];
  const float* b    = (const float*)d_in[2];
  const float* cent = (const float*)d_in[3];
  float* out = (float*)d_out;
  const int N = in_sizes[0] / (C_DIM * P_TOT);  // 128 images
  hipLaunchKernelGGL(netvlad_kernel, dim3(N), dim3(256), 0, stream,
                     x, w, b, cent, out);
}
